// MoELayer_20761871909700
// MI455X (gfx1250) — compile-verified, matrix operand order checked by reading, not run on hardware
//
#include <hip/hip_runtime.h>
#include <hip/hip_bf16.h>

// ---------------- problem constants (match reference) ----------------
#define NB   2
#define NS   2048
#define NTOK (NB * NS)     // 4096 tokens
#define DDIM 768
#define NEXP 8
#define FDIM 3072

// ---------------- GEMM tiling ----------------
// Block = 256 threads = 8 waves arranged 4(M) x 2(N).
// N tile fixed at 128 (each wave owns a 64-col slab = 4 x 16 subtiles).
// M tile TMB templated: 128 (up: wave slab 32 rows, MSUB=2, 8 WMMA/step)
//                        64 (down: wave slab 16 rows, MSUB=1, 4 WMMA/step)
#define TN 128
#define TK 32
#define NSUB 4
#define LDS_STRIDE (TK + 8)          // 40 bf16 = 80 B rows: 16B-aligned chunks, bank-spread

typedef __attribute__((ext_vector_type(16))) __bf16 v16bf;
typedef __attribute__((ext_vector_type(8)))  __bf16 v8bf;
typedef __attribute__((ext_vector_type(2)))  __bf16 v2bf;
typedef __attribute__((ext_vector_type(8)))  float  v8f;

// ---------------- workspace layout (bytes) ----------------
#define WS_COUNTS   0
#define WS_TOKLIST  128
#define WS_PROBS    (WS_TOKLIST + 4 * NEXP * NTOK)
#define WS_H        (WS_PROBS + 4 * NTOK + 128)

// =====================================================================
__global__ void zero_counts_kernel(int* counts) {
    if (threadIdx.x < NEXP) counts[threadIdx.x] = 0;
}

// ---------------- router: one wave32 per token ----------------
__global__ void router_kernel(const float* __restrict__ hidden,
                              const float* __restrict__ Wr,
                              float* __restrict__ logits_out,
                              int*   __restrict__ idx_out,
                              float* __restrict__ probs,
                              int*   __restrict__ counts,
                              int*   __restrict__ toklist) {
    const int lane  = threadIdx.x & 31;
    const int wave  = threadIdx.x >> 5;
    const int token = blockIdx.x * 8 + wave;
    if (token >= NTOK) return;

    float acc[NEXP];
#pragma unroll
    for (int e = 0; e < NEXP; ++e) acc[e] = 0.0f;

    const float* xrow = hidden + (size_t)token * DDIM;
    for (int k = lane; k < DDIM; k += 32) {
        const float x = xrow[k];
        const float* wr = Wr + (size_t)k * NEXP;
#pragma unroll
        for (int e = 0; e < NEXP; ++e) acc[e] += x * wr[e];
    }
#pragma unroll
    for (int off = 16; off > 0; off >>= 1) {
#pragma unroll
        for (int e = 0; e < NEXP; ++e) acc[e] += __shfl_down(acc[e], off, 32);
    }

    if (lane == 0) {
        float mx = acc[0];
        int amax = 0;
#pragma unroll
        for (int e = 1; e < NEXP; ++e)
            if (acc[e] > mx) { mx = acc[e]; amax = e; }
        float sum = 0.0f;
#pragma unroll
        for (int e = 0; e < NEXP; ++e) sum += __expf(acc[e] - mx);
        const float prob = 1.0f / sum;

#pragma unroll
        for (int e = 0; e < NEXP; ++e) logits_out[(size_t)token * NEXP + e] = acc[e];
        idx_out[token] = amax;
        probs[token]   = prob;
        const int pos = atomicAdd(&counts[amax], 1);
        toklist[amax * NTOK + pos] = token;
    }
}

// ---------------- fragment read: two 16B chunks per lane (ISA 7.12.2) ----------------
__device__ __forceinline__ v16bf frag_from_lds(const __bf16* base, int row, int kSel) {
    const __bf16* r = base + row * LDS_STRIDE;
    v8bf lo = *(const v8bf*)(r + kSel);        // K = kSel+0..7
    v8bf hi = *(const v8bf*)(r + kSel + 16);   // K = kSel+16..23
    return __builtin_shufflevector(lo, hi, 0,1,2,3,4,5,6,7,8,9,10,11,12,13,14,15);
}

// =====================================================================
// Grouped-GEMM body shared by up/down projections.
//   IS_UP = 1: A = fp32 hidden rows (gathered), epilogue relu(+b1) -> bf16 H
//   IS_UP = 0: A = bf16 H rows (gathered),     epilogue (+b2)*prob -> fp32 out
//   TMB: block M tile (multiple of 64); wave M slab = TMB/4; MSUB = TMB/64
template <int IS_UP, int TMB>
__global__ void ffn_gemm_kernel(const float*  __restrict__ Afp32,   // hidden (up)
                                const __bf16* __restrict__ Abf16,   // H (down)
                                const float*  __restrict__ W,       // W1 or W2
                                const float*  __restrict__ bias,    // b1 or b2
                                const float*  __restrict__ probs,
                                const int*    __restrict__ counts,
                                const int*    __restrict__ toklist,
                                __bf16* __restrict__ Hout,          // H (up)
                                float*  __restrict__ out)           // out (down)
{
    constexpr int KDIM = IS_UP ? DDIM : FDIM;   // reduction length
    constexpr int NDIM = IS_UP ? FDIM : DDIM;   // output width
    constexpr int MSUB = TMB / 64;              // 16-row subtiles per wave
    constexpr int AREP = TMB / 64;              // A staging rows-loops (64 rows per rep)

    const int e  = blockIdx.z;
    const int m0 = blockIdx.y * TMB;
    const int n0 = blockIdx.x * TN;
    const int cnt = counts[e];
    if (m0 >= cnt) return;

    const int* toks = toklist + e * NTOK;
    const float* We = W + (size_t)e * DDIM * FDIM;   // D*F == F*D

    __shared__ __align__(16) __bf16 As[TMB * LDS_STRIDE];
    __shared__ __align__(16) __bf16 Bs[TN * LDS_STRIDE];

    const int tid  = threadIdx.x;
    const int lane = tid & 31;
    const int wave = tid >> 5;
    const int mwave = wave >> 1;     // 0..3 : (TMB/4)-row slab
    const int nwave = wave & 1;      // 0..1 : 64-col slab

    // ---- A staging: AREP reps of (64 rows x 32 k, 8 elems/thread) ----
    const int rowA  = tid >> 2;          // 0..63
    const int colA8 = (tid & 3) * 8;     // 0,8,16,24
    int tokA[AREP];
#pragma unroll
    for (int a = 0; a < AREP; ++a) {
        const int mIdx = m0 + a * 64 + rowA;
        tokA[a] = (mIdx < cnt) ? toks[mIdx] : -1;
    }

    v8f acc[MSUB][NSUB];
#pragma unroll
    for (int ms = 0; ms < MSUB; ++ms)
#pragma unroll
        for (int s = 0; s < NSUB; ++s) acc[ms][s] = (v8f){};

    const int lh   = lane & 15;
    const int kSel = (lane >> 4) * 8;

    for (int kk = 0; kk < KDIM; kk += TK) {
        // ---------------- stage A (gathered token rows -> bf16, 16B stores) ----------------
#pragma unroll
        for (int a = 0; a < AREP; ++a) {
            v8bf av;
            if (tokA[a] >= 0) {
                if (IS_UP) {
                    const float4 v0 = *(const float4*)(Afp32 + (size_t)tokA[a] * KDIM + kk + colA8);
                    const float4 v1 = *(const float4*)(Afp32 + (size_t)tokA[a] * KDIM + kk + colA8 + 4);
                    av[0] = (__bf16)v0.x; av[1] = (__bf16)v0.y; av[2] = (__bf16)v0.z; av[3] = (__bf16)v0.w;
                    av[4] = (__bf16)v1.x; av[5] = (__bf16)v1.y; av[6] = (__bf16)v1.z; av[7] = (__bf16)v1.w;
                } else {
                    av = *(const v8bf*)(Abf16 + (size_t)tokA[a] * KDIM + kk + colA8);
                }
            } else {
                av = (v8bf){};
            }
            *(v8bf*)&As[(a * 64 + rowA) * LDS_STRIDE + colA8] = av;
        }
        // ---------------- stage B transposed: Bs[n][k] <- We[kk+k][n0+n] ----------------
        // bf16 k-pairs stored as 32-bit DS stores.
#pragma unroll
        for (int rep = 0; rep < 2; ++rep) {
            const int w  = tid + rep * 256;      // 0..511
            const int kp = w >> 5;               // 0..15
            const int n4 = (w & 31) * 4;         // 0..124
            const float4 a0 = *(const float4*)(We + (size_t)(kk + 2 * kp    ) * NDIM + n0 + n4);
            const float4 a1 = *(const float4*)(We + (size_t)(kk + 2 * kp + 1) * NDIM + n0 + n4);
            const float a0v[4] = {a0.x, a0.y, a0.z, a0.w};
            const float a1v[4] = {a1.x, a1.y, a1.z, a1.w};
#pragma unroll
            for (int j = 0; j < 4; ++j) {
                v2bf p;
                p[0] = (__bf16)a0v[j];
                p[1] = (__bf16)a1v[j];
                *(v2bf*)&Bs[(n4 + j) * LDS_STRIDE + 2 * kp] = p;
            }
        }
        __syncthreads();

        // ---------------- compute: MSUB A-frags x 4 B-frags ----------------
        v16bf afrag[MSUB];
#pragma unroll
        for (int ms = 0; ms < MSUB; ++ms)
            afrag[ms] = frag_from_lds(As, (mwave * MSUB + ms) * 16 + lh, kSel);
#pragma unroll
        for (int s = 0; s < NSUB; ++s) {
            const v16bf b = frag_from_lds(Bs, nwave * 64 + s * 16 + lh, kSel);
#pragma unroll
            for (int ms = 0; ms < MSUB; ++ms) {
                acc[ms][s] = __builtin_amdgcn_wmma_f32_16x16x32_bf16(false, afrag[ms], false, b,
                                                                     (short)0, acc[ms][s], false, false);
            }
        }
        __syncthreads();
    }

    // ---------------- epilogue ----------------
    const int nD  = lane & 15;
    const int mHi = (lane >> 4) << 3;
#pragma unroll
    for (int ms = 0; ms < MSUB; ++ms) {
#pragma unroll
        for (int s = 0; s < NSUB; ++s) {
            const int gn = n0 + nwave * 64 + s * 16 + nD;
#pragma unroll
            for (int r = 0; r < 8; ++r) {
                const int localm = (mwave * MSUB + ms) * 16 + r + mHi;
                if (m0 + localm < cnt) {
                    const int tok = toks[m0 + localm];
                    if (IS_UP) {
                        float v = acc[ms][s][r] + bias[e * FDIM + gn];
                        v = fmaxf(v, 0.0f);
                        Hout[(size_t)tok * FDIM + gn] = (__bf16)v;
                    } else {
                        const float v = (acc[ms][s][r] + bias[e * DDIM + gn]) * probs[tok];
                        out[(size_t)tok * DDIM + gn] = v;
                    }
                }
            }
        }
    }
}

// =====================================================================
extern "C" void kernel_launch(void* const* d_in, const int* in_sizes, int n_in,
                              void* d_out, int out_size, void* d_ws, size_t ws_size,
                              hipStream_t stream) {
    const float* hidden = (const float*)d_in[0];   // [B,S,D]
    const float* Wr     = (const float*)d_in[1];   // [D,E]
    const float* W1     = (const float*)d_in[2];   // [E,D,F]
    const float* b1     = (const float*)d_in[3];   // [E,F]
    const float* W2     = (const float*)d_in[4];   // [E,F,D]
    const float* b2     = (const float*)d_in[5];   // [E,D]

    float* out        = (float*)d_out;                            // [B,S,D]
    float* logits_out = out + (size_t)NTOK * DDIM;                // [B,S,E]
    int*   idx_out    = (int*)(logits_out + (size_t)NTOK * NEXP); // [B,S] int32

    char* ws = (char*)d_ws;
    int*    counts  = (int*)(ws + WS_COUNTS);
    int*    toklist = (int*)(ws + WS_TOKLIST);
    float*  probs   = (float*)(ws + WS_PROBS);
    __bf16* H       = (__bf16*)(ws + WS_H);

    zero_counts_kernel<<<1, 64, 0, stream>>>(counts);

    router_kernel<<<NTOK / 8, 256, 0, stream>>>(hidden, Wr, logits_out, idx_out,
                                                probs, counts, toklist);

    // Up: 128x128 block tile, 8 WMMA / wave / K-step. Grid (24, 32, 8).
    dim3 gridUp(FDIM / TN, NTOK / 128, NEXP);
    ffn_gemm_kernel<1, 128><<<gridUp, 256, 0, stream>>>(hidden, nullptr, W1, b1, nullptr,
                                                        counts, toklist, H, nullptr);

    // Down: 64x128 block tile (more M-blocks since N is only 768/128 = 6). Grid (6, 64, 8).
    dim3 gridDn(DDIM / TN, NTOK / 64, NEXP);
    ffn_gemm_kernel<0, 64><<<gridDn, 256, 0, stream>>>(nullptr, H, W2, b2, probs,
                                                       counts, toklist, nullptr, out);
}
